// CausalSelfAttention_695784702352
// MI455X (gfx1250) — matrix operand order repaired and checked by reading, not compiled, not run on hardware
//
#include <hip/hip_runtime.h>
#include <math.h>

#define DIM   2048
#define NHEAD 16
#define HD    128
#define SEQ   2048
#define BATCH 2
#define MROWS (BATCH*SEQ)   // 4096
#define NQKV  (3*DIM)       // 6144

typedef __attribute__((ext_vector_type(16))) __bf16 v16bf;
typedef __attribute__((ext_vector_type(8)))  float  v8f;

__device__ __forceinline__ unsigned short f2bf(float f) {
  unsigned int u = __builtin_bit_cast(unsigned int, f);
  u += 0x7FFFu + ((u >> 16) & 1u);   // round-to-nearest-even
  return (unsigned short)(u >> 16);
}
__device__ __forceinline__ float bf2f(unsigned short b) {
  unsigned int u = ((unsigned int)b) << 16;
  return __builtin_bit_cast(float, u);
}
__device__ __forceinline__ v8f wmma_bf16(v16bf a, v16bf b, v8f c) {
  return __builtin_amdgcn_wmma_f32_16x16x32_bf16(false, a, false, b, (short)0, c, false, false);
}
__device__ __forceinline__ v8f vzero8() {
  v8f z = {0.f,0.f,0.f,0.f,0.f,0.f,0.f,0.f};
  return z;
}

// ---------------------------------------------------------------------------
// Generic fp32-in GEMM, bf16 WMMA compute, f32 accumulate.
// C[M,N] = A[M,K] @ B[K,N].  Block tile 128x128, K-step 32, 256 thr = 8 waves.
// mode 0: store C fp32 row-major.
// mode 1: (QKV) scatter bf16 into q/k/v [B,H,S,D] buffers.
// ---------------------------------------------------------------------------
__global__ __launch_bounds__(256)
void gemm_bf16_kernel(const float* __restrict__ A, const float* __restrict__ Bm,
                      int K, int N, float* __restrict__ C,
                      unsigned short* __restrict__ qw, unsigned short* __restrict__ kw,
                      unsigned short* __restrict__ vw, int mode)
{
  // fragment-swizzled tiles: [16-row/col block 0..7][lane 0..31][half 0..15]
  __shared__ __align__(32) unsigned short sA[8*32*16];
  __shared__ __align__(32) unsigned short sB[8*32*16];

  const int t    = threadIdx.x;
  const int wave = t >> 5, lane = t & 31;
  const int m0   = blockIdx.y * 128;
  const int n0   = blockIdx.x * 128;
  const int wm   = wave >> 1;   // 0..3 : 32 rows each
  const int wn   = wave & 1;    // 0..1 : 64 cols each

  v8f acc[2][4];
#pragma unroll
  for (int i = 0; i < 2; ++i)
#pragma unroll
    for (int j = 0; j < 4; ++j) acc[i][j] = vzero8();

  const int a_k2 = (t & 15) * 2;   // even k within 32
  const int a_r0 = t >> 4;         // row 0..15 (strided by 16)
  const int b_n  = t & 127;        // col 0..127
  const int b_k0 = (t >> 7) * 2;   // 0 or 2

  for (int kb = 0; kb < K; kb += 32) {
    __syncthreads();
    // ---- stage A tile (128 x 32) fp32 -> bf16, A-frag swizzle ----
#pragma unroll
    for (int i = 0; i < 8; ++i) {
      int r = a_r0 + 16*i;
      float2 f = *(const float2*)(A + (size_t)(m0 + r)*K + kb + a_k2);
      int la = (((a_k2 >> 3) & 1) << 4) + (r & 15);
      int h  = ((a_k2 >> 4) << 3) + (a_k2 & 7);
      unsigned int pk = (unsigned int)f2bf(f.x) | ((unsigned int)f2bf(f.y) << 16);
      *(unsigned int*)&sA[(((r >> 4)*32 + la) << 4) + h] = pk;
    }
    // ---- stage B tile (32 x 128) fp32 -> bf16, B-frag swizzle ----
#pragma unroll
    for (int i = 0; i < 8; ++i) {
      int kk = b_k0 + 4*i;  // even 0..30
      float f0 = Bm[(size_t)(kb + kk    )*N + n0 + b_n];
      float f1 = Bm[(size_t)(kb + kk + 1)*N + n0 + b_n];
      int la = (((kk >> 3) & 1) << 4) + (b_n & 15);
      int h  = ((kk >> 4) << 3) + (kk & 7);
      unsigned int pk = (unsigned int)f2bf(f0) | ((unsigned int)f2bf(f1) << 16);
      *(unsigned int*)&sB[(((b_n >> 4)*32 + la) << 4) + h] = pk;
    }
    __syncthreads();

    v16bf af[2], bfr[4];
#pragma unroll
    for (int i = 0; i < 2; ++i)
      af[i] = *(const v16bf*)&sA[(((wm*2 + i)*32 + lane) << 4)];
#pragma unroll
    for (int j = 0; j < 4; ++j)
      bfr[j] = *(const v16bf*)&sB[(((wn*4 + j)*32 + lane) << 4)];
#pragma unroll
    for (int i = 0; i < 2; ++i)
#pragma unroll
      for (int j = 0; j < 4; ++j)
        acc[i][j] = wmma_bf16(af[i], bfr[j], acc[i][j]);
  }

  const int rg = (lane >> 4) * 8;
  const int cl = lane & 15;
#pragma unroll
  for (int i = 0; i < 2; ++i)
#pragma unroll
    for (int j = 0; j < 4; ++j)
#pragma unroll
      for (int vv = 0; vv < 8; ++vv) {
        int mm = m0 + wm*32 + i*16 + vv + rg;
        int nn = n0 + wn*64 + j*16 + cl;
        float val = acc[i][j][vv];
        if (mode == 0) {
          C[(size_t)mm * N + nn] = val;
        } else {
          int which = nn >> 11;          // / 2048
          int hh = (nn & 2047) >> 7;     // / 128
          int dd = nn & 127;
          int b  = mm >> 11;             // / SEQ
          int s  = mm & 2047;
          size_t idx = (((size_t)b*NHEAD + hh)*SEQ + s)*HD + dd;
          unsigned short* dst = (which == 0) ? qw : ((which == 1) ? kw : vw);
          dst[idx] = f2bf(val);
        }
      }
}

// ---------------------------------------------------------------------------
// RoPE in place on bf16 Q/K, layout [B,H,S,D]; one thread = one (row, i) pair.
// ---------------------------------------------------------------------------
__global__ __launch_bounds__(256)
void rope_kernel(unsigned short* __restrict__ qw, unsigned short* __restrict__ kw)
{
  size_t idx = (size_t)blockIdx.x * blockDim.x + threadIdx.x;
  int    i   = (int)(idx & 63);
  size_t row = idx >> 6;                 // (b*H + h)*S + s
  int    s   = (int)(row & (SEQ - 1));
  size_t base = row * HD;
  float ang = (float)s * __powf(10000.0f, -(float)i * (1.0f/64.0f));
  float sn, c;
  __sincosf(ang, &sn, &c);
  {
    float x1 = bf2f(qw[base + i]), x2 = bf2f(qw[base + i + 64]);
    qw[base + i]      = f2bf(x1*c - x2*sn);
    qw[base + i + 64] = f2bf(x2*c + x1*sn);
  }
  {
    float x1 = bf2f(kw[base + i]), x2 = bf2f(kw[base + i + 64]);
    kw[base + i]      = f2bf(x1*c - x2*sn);
    kw[base + i + 64] = f2bf(x2*c + x1*sn);
  }
}

// ---------------------------------------------------------------------------
// Flash attention: block = 4 waves = 64 query rows of one (b,h).
// Streams 64-key causal blocks; WMMA for QK^T and PV; streaming softmax.
// Output written to ATT [B,S,DIM] fp32.
// ---------------------------------------------------------------------------
__global__ __launch_bounds__(128)
void attn_kernel(const unsigned short* __restrict__ Qw,
                 const unsigned short* __restrict__ Kw,
                 const unsigned short* __restrict__ Vw,
                 float* __restrict__ ATT)
{
  __shared__ __align__(32) unsigned short sQ[4][4][32][16]; // wave, kchunk(D/32), lane, half
  __shared__ __align__(32) unsigned short sK[4][4][32][16]; // kchunk(D/32), jtile, lane, half
  __shared__ __align__(32) unsigned short sV[2][8][32][16]; // kchunk(64keys/32), dtile, lane, half
  __shared__ __align__(32) unsigned short sP[4][2][32][16]; // wave, kchunk, lane, half

  const int t    = threadIdx.x;
  const int wave = t >> 5, lane = t & 31;
  const int qblk = blockIdx.x;
  const int bh   = blockIdx.y;
  const size_t headBase = (size_t)bh * SEQ * HD;
  const int q0 = qblk * 64;

  // stage Q (64 x 128) as A-fragments
  for (int e = t; e < 64*128; e += 128) {
    int r = e >> 7, d = e & 127;
    unsigned short val = Qw[headBase + (size_t)(q0 + r)*HD + d];
    int k32 = d & 31;
    sQ[r >> 4][d >> 5][(((k32 >> 3) & 1) << 4) + (r & 15)]
      [((k32 >> 4) << 3) + (k32 & 7)] = val;
  }
  __syncthreads();
  v16bf qa[4];
#pragma unroll
  for (int c = 0; c < 4; ++c) qa[c] = *(const v16bf*)&sQ[wave][c][lane][0];

  v8f o[8];
#pragma unroll
  for (int dt = 0; dt < 8; ++dt) o[dt] = vzero8();
  float mrow[8], lrow[8];
#pragma unroll
  for (int vv = 0; vv < 8; ++vv) { mrow[vv] = -3.0e38f; lrow[vv] = 0.f; }

  const float scale = 0.08838834764831845f;  // 1/sqrt(128)
  const int rg = (lane >> 4) * 8;
  const int cl = lane & 15;

  for (int kblk = 0; kblk <= qblk; ++kblk) {
    const int k0 = kblk * 64;
    __syncthreads();
    // stage K (as K^T B-frags) and V (as B-frags)
    for (int e = t; e < 64*128; e += 128) {
      int r = e >> 7, d = e & 127;
      unsigned short kv = Kw[headBase + (size_t)(k0 + r)*HD + d];
      sK[d >> 5][r >> 4][(((d >> 3) & 1) << 4) + (r & 15)][(((d >> 4) & 1) << 3) + (d & 7)] = kv;
      unsigned short vv2 = Vw[headBase + (size_t)(k0 + r)*HD + d];
      sV[r >> 5][d >> 4][(((r >> 3) & 1) << 4) + (d & 15)][(((r >> 4) & 1) << 3) + (r & 7)] = vv2;
    }
    __syncthreads();

    // scores: 16 (q rows) x 64 (keys), 4 jtiles x 4 D-chunks
    v8f sc[4];
#pragma unroll
    for (int jt = 0; jt < 4; ++jt) sc[jt] = vzero8();
#pragma unroll
    for (int c = 0; c < 4; ++c)
#pragma unroll
      for (int jt = 0; jt < 4; ++jt)
        sc[jt] = wmma_bf16(qa[c], *(const v16bf*)&sK[c][jt][lane][0], sc[jt]);

    // scale + causal mask
    const int qbase = q0 + wave*16 + rg;
#pragma unroll
    for (int jt = 0; jt < 4; ++jt)
#pragma unroll
      for (int vv = 0; vv < 8; ++vv) {
        float x = sc[jt][vv] * scale;
        if (k0 + jt*16 + cl > qbase + vv) x = -3.0e38f;
        sc[jt][vv] = x;
      }

    // streaming softmax
    float rmax[8];
#pragma unroll
    for (int vv = 0; vv < 8; ++vv) {
      float x = fmaxf(fmaxf(sc[0][vv], sc[1][vv]), fmaxf(sc[2][vv], sc[3][vv]));
#pragma unroll
      for (int off = 8; off >= 1; off >>= 1) x = fmaxf(x, __shfl_xor(x, off, 32));
      rmax[vv] = x;
    }
#pragma unroll
    for (int vv = 0; vv < 8; ++vv) {
      float mnew  = fmaxf(mrow[vv], rmax[vv]);
      float alpha = __expf(mrow[vv] - mnew);
      mrow[vv] = mnew;
      lrow[vv] *= alpha;
#pragma unroll
      for (int dt = 0; dt < 8; ++dt) o[dt][vv] *= alpha;
      float ps = 0.f;
#pragma unroll
      for (int jt = 0; jt < 4; ++jt) {
        float p = __expf(sc[jt][vv] - mnew);
        ps += p;
        int jj  = jt*16 + cl;
        int k32 = jj & 31;
        sP[wave][jj >> 5][(((k32 >> 3) & 1) << 4) + (vv + rg)]
          [((k32 >> 4) << 3) + (k32 & 7)] = f2bf(p);
      }
#pragma unroll
      for (int off = 8; off >= 1; off >>= 1) ps += __shfl_xor(ps, off, 32);
      lrow[vv] += ps;
    }

    // wave-private LDS round trip: C-layout P -> A-fragment layout
    asm volatile("s_wait_dscnt 0x0" ::: "memory");
    v16bf pa0 = *(const v16bf*)&sP[wave][0][lane][0];
    v16bf pa1 = *(const v16bf*)&sP[wave][1][lane][0];
#pragma unroll
    for (int dt = 0; dt < 8; ++dt) {
      o[dt] = wmma_bf16(pa0, *(const v16bf*)&sV[0][dt][lane][0], o[dt]);
      o[dt] = wmma_bf16(pa1, *(const v16bf*)&sV[1][dt][lane][0], o[dt]);
    }
  }

  const int b  = bh >> 4;   // / NHEAD
  const int hh = bh & 15;
#pragma unroll
  for (int dt = 0; dt < 8; ++dt)
#pragma unroll
    for (int vv = 0; vv < 8; ++vv) {
      int s   = q0 + wave*16 + rg + vv;
      int col = hh*HD + dt*16 + cl;
      ATT[((size_t)b*SEQ + s)*DIM + col] = o[dt][vv] / lrow[vv];
    }
}

// ---------------------------------------------------------------------------
extern "C" void kernel_launch(void* const* d_in, const int* in_sizes, int n_in,
                              void* d_out, int out_size, void* d_ws, size_t ws_size,
                              hipStream_t stream)
{
  (void)in_sizes; (void)n_in; (void)out_size; (void)ws_size;
  const float* x     = (const float*)d_in[0];
  const float* w_qkv = (const float*)d_in[1];
  const float* w_out = (const float*)d_in[2];
  // d_in[3] (mask) is the causal tril; handled analytically.
  float* out = (float*)d_out;

  const size_t nQ = (size_t)BATCH * NHEAD * SEQ * HD;       // 8,388,608
  unsigned short* Qb = (unsigned short*)d_ws;
  unsigned short* Kb = Qb + nQ;
  unsigned short* Vb = Kb + nQ;
  float* ATT = (float*)(Vb + nQ);                           // 32 MB fp32

  dim3 g1(NQKV / 128, MROWS / 128);                         // 48 x 32
  gemm_bf16_kernel<<<g1, 256, 0, stream>>>(x, w_qkv, DIM, NQKV,
                                           nullptr, Qb, Kb, Vb, 1);

  int ropeThreads = BATCH * NHEAD * SEQ * 64;               // 4,194,304
  rope_kernel<<<ropeThreads / 256, 256, 0, stream>>>(Qb, Kb);

  dim3 g2(SEQ / 64, BATCH * NHEAD);                         // 32 x 32
  attn_kernel<<<g2, 128, 0, stream>>>(Qb, Kb, Vb, ATT);

  dim3 g3(DIM / 128, MROWS / 128);                          // 16 x 32
  gemm_bf16_kernel<<<g3, 256, 0, stream>>>(ATT, w_out, DIM, DIM,
                                           out, nullptr, nullptr, nullptr, 0);
}
